// GAT_3075196584311
// MI455X (gfx1250) — matrix operand order repaired and checked by reading, not compiled, not run on hardware
//
#include <hip/hip_runtime.h>
#include <cmath>

typedef __attribute__((ext_vector_type(16))) _Float16 v16h;
typedef __attribute__((ext_vector_type(8)))  _Float16 v8h;
typedef __attribute__((ext_vector_type(8)))  float    v8f;
typedef __attribute__((ext_vector_type(8)))  float    v8fv;

#define NEG_SLOPE 0.2f

// ---------------------------------------------------------------- utilities

__global__ void fill_f32(float* __restrict__ p, float v, long n) {
    long i = (long)blockIdx.x * blockDim.x + threadIdx.x;
    if (i < n) p[i] = v;
}

// W: [K, C] row-major f32  ->  Wt: [C, K] f16 (column-major of W)
__global__ void convert_wt(const float* __restrict__ W, _Float16* __restrict__ Wt,
                           int K, int C) {
    int i = blockIdx.x * blockDim.x + threadIdx.x;
    if (i < K * C) {
        int k = i / C, c = i % C;
        Wt[(long)c * K + k] = (_Float16)W[i];
    }
}

__device__ __forceinline__ float atomicMaxF(float* addr, float v) {
    if (v >= 0.f)
        return __int_as_float(atomicMax((int*)addr, __float_as_int(v)));
    return __uint_as_float(atomicMin((unsigned int*)addr, __float_as_uint(v)));
}

__device__ __forceinline__ float lrelu(float x) { return x > 0.f ? x : NEG_SLOPE * x; }

// ---------------------------------------------------------------- WMMA GEMM
// C[M,Ncols] = A[M,K](f32) @ B[K,Ncols], with Bt = B^T in f16 ([Ncols,K]).
// One wave computes a 16 x (16*CT) strip: one A fragment feeds CT WMMAs.
// A frag: lane holds row lane%16, K = kb+{0..7} and kb+{16..23}, kb = 8*(lane/16).
// B frag: lane holds col lane%16, K = kb+{0..7} and kb+{16..23}, kb = 8*(lane/16).
template <int CT>
__global__ void gemm_wmma(const float* __restrict__ A, const _Float16* __restrict__ Bt,
                          float* __restrict__ C, int M, int Ncols, int K) {
    const int wave = threadIdx.x >> 5;
    const int lane = threadIdx.x & 31;
    const int rowTile = blockIdx.x * 8 + wave;
    const int nRowTiles = M >> 4;
    if (rowTile >= nRowTiles) return;                 // uniform per-wave guard
    const int colBase = blockIdx.y * CT;              // in 16-col tiles
    const int hi = lane >> 4;                         // 0 | 1 (half-wave)
    const int l16 = lane & 15;

    const float* arow = A + (long)((rowTile << 4) + l16) * K + (hi << 3);
    const _Float16* brow[CT];
#pragma unroll
    for (int t = 0; t < CT; ++t)
        brow[t] = Bt + (long)(((colBase + t) << 4) + l16) * K + (hi << 3);

    v8f acc[CT] = {};
    for (int k = 0; k < K; k += 32) {
        v8fv fa0 = *(const v8fv*)(arow + k);
        v8fv fa1 = *(const v8fv*)(arow + k + 16);
        v8h  ha0 = __builtin_convertvector(fa0, v8h);
        v8h  ha1 = __builtin_convertvector(fa1, v8h);
        v16h af  = __builtin_shufflevector(ha0, ha1,
                     0,1,2,3,4,5,6,7,8,9,10,11,12,13,14,15);
#pragma unroll
        for (int t = 0; t < CT; ++t) {
            v8h  hb0 = *(const v8h*)(brow[t] + k);
            v8h  hb1 = *(const v8h*)(brow[t] + k + 16);
            v16h bf  = __builtin_shufflevector(hb0, hb1,
                         0,1,2,3,4,5,6,7,8,9,10,11,12,13,14,15);
            acc[t] = __builtin_amdgcn_wmma_f32_16x16x32_f16(
                         false, af, false, bf, (short)0, acc[t], false, false);
        }
    }
    // D layout: VGPR v -> row = 8*hi + v, col = lane&15
    float* crow = C + (long)((rowTile << 4) + (hi << 3)) * Ncols + l16;
#pragma unroll
    for (int t = 0; t < CT; ++t) {
        float* cp = crow + ((colBase + t) << 4);
#pragma unroll
        for (int v = 0; v < 8; ++v) cp[(long)v * Ncols] = acc[t][v];
    }
}

// ---------------------------------------------------------------- attention scalars
// One wave per node: as[n,h] = dot(H[n, h*64: ], a_src[h*64:]), same for dst.
template <int NH>
__global__ void alpha_kernel(const float* __restrict__ H,
                             const float* __restrict__ a_src,
                             const float* __restrict__ a_dst,
                             float* __restrict__ as_, float* __restrict__ ad_, int n) {
    const int wave = threadIdx.x >> 5, lane = threadIdx.x & 31;
    const int node = blockIdx.x * 8 + wave;
    if (node >= n) return;
    const float* hp = H + (long)node * (NH * 64);
    float ps[NH], pd[NH];
#pragma unroll
    for (int h = 0; h < NH; ++h) { ps[h] = 0.f; pd[h] = 0.f; }
#pragma unroll
    for (int j = 0; j < NH * 2; ++j) {
        const int c = lane + 32 * j;
        const int h = j >> 1;                 // uniform per j
        const float v = hp[c];
        ps[h] += v * a_src[c];
        pd[h] += v * a_dst[c];
    }
#pragma unroll
    for (int h = 0; h < NH; ++h) {
        float s = ps[h], d = pd[h];
#pragma unroll
        for (int off = 16; off; off >>= 1) {
            s += __shfl_xor(s, off, 32);
            d += __shfl_xor(d, off, 32);
        }
        if (lane == 0) { as_[(long)node * NH + h] = s; ad_[(long)node * NH + h] = d; }
    }
}

// ---------------------------------------------------------------- edge passes
__device__ __forceinline__ void edge_sd(const int* adj, int E, int e, int& s, int& d) {
    if (e < E) { s = adj[e]; d = adj[E + e]; }
    else       { s = e - E;  d = e - E; }        // appended self loops
}

template <int NH>
__global__ void edge_max(const int* __restrict__ adj, int E, int n,
                         const float* __restrict__ as_, const float* __restrict__ ad_,
                         float* __restrict__ m) {
    int e = blockIdx.x * blockDim.x + threadIdx.x;
    if (e >= E + n) return;
    int s, d; edge_sd(adj, E, e, s, d);
#pragma unroll
    for (int h = 0; h < NH; ++h) {
        float a = lrelu(as_[(long)s * NH + h] + ad_[(long)d * NH + h]);
        atomicMaxF(&m[(long)d * NH + h], a);
    }
}

template <int NH>
__global__ void edge_exp(const int* __restrict__ adj, int E, int n,
                         const float* __restrict__ as_, const float* __restrict__ ad_,
                         const float* __restrict__ m, float* __restrict__ ebuf,
                         float* __restrict__ denom) {
    int e = blockIdx.x * blockDim.x + threadIdx.x;
    if (e >= E + n) return;
    int s, d; edge_sd(adj, E, e, s, d);
#pragma unroll
    for (int h = 0; h < NH; ++h) {
        float a  = lrelu(as_[(long)s * NH + h] + ad_[(long)d * NH + h]);
        float ev = __expf(a - m[(long)d * NH + h]);
        ebuf[(long)e * NH + h] = ev;
        atomicAdd(&denom[(long)d * NH + h], ev);
    }
}

// One wave per edge: acc[dst, :] += H[src, :] * softmax_weight(head(c)).
// e (and thus s, d) are wave-uniform: force them scalar so the weight loads
// and address math go down the SALU/SMEM path.
template <int C>
__global__ void edge_msg(const int* __restrict__ adj, int E, int n,
                         const float* __restrict__ H, const float* __restrict__ ebuf,
                         const float* __restrict__ denom, float* __restrict__ acc) {
    constexpr int NH = C / 64;
    const int wave = threadIdx.x >> 5, lane = threadIdx.x & 31;
    const int e = blockIdx.x * 8 + wave;
    if (e >= E + n) return;
    int s, d; edge_sd(adj, E, e, s, d);
    s = __builtin_amdgcn_readfirstlane(s);
    d = __builtin_amdgcn_readfirstlane(d);
    const float* hs = H   + (long)s * C;
    float*       ap = acc + (long)d * C;
    float w[NH];
#pragma unroll
    for (int h = 0; h < NH; ++h)
        w[h] = ebuf[(long)e * NH + h] / denom[(long)d * NH + h];
#pragma unroll
    for (int j = 0; j < C / 32; ++j) {
        const int c = lane + 32 * j;
        atomicAdd(&ap[c], hs[c] * w[j >> 1]);   // head uniform per 32-chan slice
    }
}

__global__ void elu_bias(float* __restrict__ acc, const float* __restrict__ b, long total) {
    long i = (long)blockIdx.x * blockDim.x + threadIdx.x;
    if (i < total) {
        float v = acc[i] + b[i & 255];
        acc[i] = v > 0.f ? v : __expf(v) - 1.f;
    }
}

// One wave per node: out[n,:] = log_softmax(acc[n,:]+b, 64)
__global__ void logsoftmax64(const float* __restrict__ acc, const float* __restrict__ b,
                             float* __restrict__ out, int n) {
    const int wave = threadIdx.x >> 5, lane = threadIdx.x & 31;
    const int node = blockIdx.x * 8 + wave;
    if (node >= n) return;
    const float* p = acc + (long)node * 64;
    float v0 = p[lane] + b[lane];
    float v1 = p[lane + 32] + b[lane + 32];
    float mx = fmaxf(v0, v1);
#pragma unroll
    for (int off = 16; off; off >>= 1) mx = fmaxf(mx, __shfl_xor(mx, off, 32));
    float s = __expf(v0 - mx) + __expf(v1 - mx);
#pragma unroll
    for (int off = 16; off; off >>= 1) s += __shfl_xor(s, off, 32);
    float lse = mx + __logf(s);
    out[(long)node * 64 + lane]      = v0 - lse;
    out[(long)node * 64 + lane + 32] = v1 - lse;
}

// ---------------------------------------------------------------- launch
extern "C" void kernel_launch(void* const* d_in, const int* in_sizes, int n_in,
                              void* d_out, int out_size, void* d_ws, size_t ws_size,
                              hipStream_t stream) {
    const float* x        = (const float*)d_in[0];
    const int*   adj      = (const int*)  d_in[1];
    const float* W1       = (const float*)d_in[2];
    const float* att_src1 = (const float*)d_in[3];
    const float* att_dst1 = (const float*)d_in[4];
    const float* b1       = (const float*)d_in[5];
    const float* W2       = (const float*)d_in[6];
    const float* att_src2 = (const float*)d_in[7];
    const float* att_dst2 = (const float*)d_in[8];
    const float* b2       = (const float*)d_in[9];
    float* out = (float*)d_out;

    const int N = in_sizes[0] / 256;       // 50000
    const int E = in_sizes[1] / 2;         // 800000
    const int TOT = E + N;
    const int K = 256;

    // ---- workspace layout (256B aligned slabs) ----
    char* base = (char*)d_ws;
    size_t off = 0;
    auto take = [&](size_t bytes) -> char* {
        off = (off + 255) & ~(size_t)255;
        char* p = base + off;
        off += bytes;
        return p;
    };
    _Float16* w1t  = (_Float16*)take((size_t)256 * 256 * 2);
    _Float16* w2t  = (_Float16*)take((size_t)64  * 256 * 2);
    float* acc1 = (float*)take((size_t)N * 256 * 4);
    float* as1  = (float*)take((size_t)N * 4 * 4);
    float* ad1  = (float*)take((size_t)N * 4 * 4);
    float* m1   = (float*)take((size_t)N * 4 * 4);
    float* d1   = (float*)take((size_t)N * 4 * 4);
    float* e1   = (float*)take((size_t)TOT * 4 * 4);
    char*  hreg = take((size_t)N * 256 * 4);           // h1; reused after layer-1 scatter
    float* h1   = (float*)hreg;
    // overlay onto dead h1 region for layer 2 (29.8 MB < 51.2 MB):
    size_t o2 = 0;
    auto take2 = [&](size_t bytes) -> char* {
        o2 = (o2 + 255) & ~(size_t)255;
        char* p = hreg + o2;
        o2 += bytes;
        return p;
    };
    float* g    = (float*)take2((size_t)N * 64 * 4);
    float* acc2 = (float*)take2((size_t)N * 64 * 4);
    float* e2   = (float*)take2((size_t)TOT * 4);
    float* as2  = (float*)take2((size_t)N * 4);
    float* ad2  = (float*)take2((size_t)N * 4);
    float* m2   = (float*)take2((size_t)N * 4);
    float* d2   = (float*)take2((size_t)N * 4);

    const float NEG_INF = -HUGE_VALF;
    auto cdiv = [](long a, long b) { return (unsigned)((a + b - 1) / b); };

    // ---- prep: W -> f16 transposed ----
    convert_wt<<<cdiv(256 * 256, 256), 256, 0, stream>>>(W1, w1t, 256, 256);
    convert_wt<<<cdiv(256 * 64, 256), 256, 0, stream>>>(W2, w2t, 256, 64);

    // ---- layer 1 ----
    fill_f32<<<cdiv((long)N * 4, 256), 256, 0, stream>>>(m1, NEG_INF, (long)N * 4);
    fill_f32<<<cdiv((long)N * 4, 256), 256, 0, stream>>>(d1, 0.f, (long)N * 4);
    fill_f32<<<cdiv((long)N * 256, 256), 256, 0, stream>>>(acc1, 0.f, (long)N * 256);

    gemm_wmma<4><<<dim3(cdiv(N / 16, 8), 4), 256, 0, stream>>>(x, w1t, h1, N, 256, K);
    alpha_kernel<4><<<cdiv(N, 8), 256, 0, stream>>>(h1, att_src1, att_dst1, as1, ad1, N);
    edge_max<4><<<cdiv(TOT, 256), 256, 0, stream>>>(adj, E, N, as1, ad1, m1);
    edge_exp<4><<<cdiv(TOT, 256), 256, 0, stream>>>(adj, E, N, as1, ad1, m1, e1, d1);
    edge_msg<256><<<cdiv(TOT, 8), 256, 0, stream>>>(adj, E, N, h1, e1, d1, acc1);
    elu_bias<<<cdiv((long)N * 256, 256), 256, 0, stream>>>(acc1, b1, (long)N * 256);

    // ---- layer 2 (h1 region now reused for g/acc2/e2/...) ----
    fill_f32<<<cdiv(N, 256), 256, 0, stream>>>(m2, NEG_INF, N);
    fill_f32<<<cdiv(N, 256), 256, 0, stream>>>(d2, 0.f, N);
    fill_f32<<<cdiv((long)N * 64, 256), 256, 0, stream>>>(acc2, 0.f, (long)N * 64);

    gemm_wmma<4><<<dim3(cdiv(N / 16, 8), 1), 256, 0, stream>>>(acc1, w2t, g, N, 64, K);
    alpha_kernel<1><<<cdiv(N, 8), 256, 0, stream>>>(g, att_src2, att_dst2, as2, ad2, N);
    edge_max<1><<<cdiv(TOT, 256), 256, 0, stream>>>(adj, E, N, as2, ad2, m2);
    edge_exp<1><<<cdiv(TOT, 256), 256, 0, stream>>>(adj, E, N, as2, ad2, m2, e2, d2);
    edge_msg<64><<<cdiv(TOT, 8), 256, 0, stream>>>(adj, E, N, g, e2, d2, acc2);

    logsoftmax64<<<cdiv(N, 8), 256, 0, stream>>>(acc2, b2, out, N);
}
